// MultiHeadAttention_58926951301353
// MI455X (gfx1250) — compile-verified, matrix operand order checked by reading, not run on hardware
//
#include <hip/hip_runtime.h>
#include <hip/hip_bf16.h>

// ---- MI455X / gfx1250 MHA: bf16 WMMA + TDM pipeline --------------------------
// B=2, S=2048, DIM=1024, H=16, dk=64.
//  * all GEMMs via v_wmma_f32_16x16x32_bf16 (f32 accumulate)
//  * GEMM waves do 32(M)x64(N) tiles: B fragments reused across 2 A tiles
//  * flash attention: 32 queries/wave, K/V chunks DMA'd to LDS by the Tensor
//    Data Mover (tensor_load_to_lds, double-buffered, s_wait_tensorcnt),
//    transposed-score layout so the softmax'd P fragment feeds the PV WMMA
//    with zero cross-lane movement.

typedef __attribute__((ext_vector_type(16))) __bf16   v16bf;
typedef __attribute__((ext_vector_type(8)))  float    v8f;
typedef __attribute__((ext_vector_type(2)))  __bf16   bf16x2;
typedef __attribute__((ext_vector_type(4)))  unsigned int u32x4;
typedef __attribute__((ext_vector_type(8)))  unsigned int u32x8;

#define BATCH 2
#define SEQ   2048
#define DIM_  1024
#define NHEAD 16
#define DKH   64
#define MTOT  (BATCH * SEQ)   // 4096 token rows

static __device__ __forceinline__ v8f wmma_bf16(v16bf a, v16bf b, v8f c) {
    return __builtin_amdgcn_wmma_f32_16x16x32_bf16(
        false, a, false, b, (short)0, c, false, false);
}

// Load one 16-bit A/B fragment (one K=32 step) from a row-major row.
// element e holds K = kb + koff + (e<8 ? e : e+8); koff = 8 for lanes 16..31.
// Works for global or LDS rows (compiler picks global_load_b128 / ds_load).
static __device__ __forceinline__ v16bf load_frag_row(const __bf16* rowp, int kb, int koff) {
    const bf16x2* r = (const bf16x2*)rowp;
    int b0 = (kb + koff) >> 1;
    int b1 = (kb + 16 + koff) >> 1;
    v16bf f;
#pragma unroll
    for (int p = 0; p < 4; ++p) {
        bf16x2 t0 = r[b0 + p];
        bf16x2 t1 = r[b1 + p];
        f[2 * p]     = t0[0];
        f[2 * p + 1] = t0[1];
        f[8 + 2 * p] = t1[0];
        f[9 + 2 * p] = t1[1];
    }
    return f;
}

// ---- Tensor Data Mover: async 2-D bf16 tile load, global -> LDS --------------
// D# per CDNA5 ISA ch.8: group0 {count=1, lds_addr, 57b global_addr, type=2},
// group1 {data_size=2B, tensor_dim0/1, tile_dim0/1, dim0_stride}. 2-group form
// (VADDR2/3 omitted -> NULL) is valid for tensors up to 2-D.
static __device__ __forceinline__ void tdm_load_2d(unsigned int lds_off, const void* gaddr,
                                                   unsigned int tensor_d0, unsigned int tensor_d1,
                                                   unsigned int tile_d0, unsigned int tile_d1,
                                                   unsigned long long stride0) {
    unsigned long long ga = (unsigned long long)gaddr;
    u32x4 g0;
    g0[0] = 1u;                                                // count=1 (user D#)
    g0[1] = lds_off;                                           // LDS byte address
    g0[2] = (unsigned int)ga;                                  // global_addr[31:0]
    g0[3] = (unsigned int)((ga >> 32) & 0x1FFFFFFu) | (2u << 30); // addr[56:32], type=2
    u32x8 g1;
    g1[0] = (1u << 16);                                        // data_size=1 -> 2 bytes
    g1[1] = (tensor_d0 & 0xFFFFu) << 16;                       // tensor_dim0[15:0]
    g1[2] = ((tensor_d0 >> 16) & 0xFFFFu) | ((tensor_d1 & 0xFFFFu) << 16);
    g1[3] = ((tensor_d1 >> 16) & 0xFFFFu) | (tile_d0 << 16);   // tile_dim0
    g1[4] = tile_d1 & 0xFFFFu;                                 // tile_dim1 (tile_dim2=0)
    g1[5] = (unsigned int)stride0;                             // dim0_stride[31:0]
    g1[6] = (unsigned int)((stride0 >> 32) & 0xFFFFu);         // dim0_stride[47:32]
    g1[7] = 0u;
    asm volatile("tensor_load_to_lds %0, %1" :: "s"(g0), "s"(g1) : "memory");
}

static __device__ __forceinline__ unsigned int lds_addr_of(const void* p) {
    // generic shared pointer = {shared aperture, LDS byte offset}; low 32b = offset
    return (unsigned int)(unsigned long long)p;
}

// ---------------------------------------------------------------- fp32 -> bf16
__global__ void cvt_f32_bf16(const float* __restrict__ src, __bf16* __restrict__ dst, int n) {
    int i = blockIdx.x * blockDim.x + threadIdx.x;
    if (i < n) dst[i] = (__bf16)src[i];
}

// ------------------------------------------------- projection GEMM (WMMA bf16)
// Y[m,n] = sum_k X[m,k]*W[n,k] + bias[n].  One wave: 32(M) x 64(N) tile.
// vmode 0: store bf16 [B,H,S,dk] (Q,K);  vmode 1: store bf16 [B,H,dk,S] (V^T)
__global__ void proj_gemm_kernel(const __bf16* __restrict__ X, const __bf16* __restrict__ W,
                                 const float* __restrict__ bias, __bf16* __restrict__ Y,
                                 int vmode) {
    int lane = threadIdx.x;
    int koff = (lane >> 4) * 8;
    int lm   = lane & 15;
    int mt   = blockIdx.x;        // 32-row M tile
    int nb   = blockIdx.y * 64;   // 64-col N block

    const __bf16* Xr0 = X + (size_t)(mt * 32 + lm) * DIM_;
    const __bf16* Xr1 = X + (size_t)(mt * 32 + 16 + lm) * DIM_;
    v8f z = {};
    v8f acc[2][4];
#pragma unroll
    for (int mi = 0; mi < 2; ++mi)
#pragma unroll
        for (int nt = 0; nt < 4; ++nt) acc[mi][nt] = z;

    for (int kb = 0; kb < DIM_; kb += 32) {
        v16bf a0 = load_frag_row(Xr0, kb, koff);
        v16bf a1 = load_frag_row(Xr1, kb, koff);
#pragma unroll
        for (int nt = 0; nt < 4; ++nt) {
            v16bf bfr = load_frag_row(W + (size_t)(nb + nt * 16 + lm) * DIM_, kb, koff);
            acc[0][nt] = wmma_bf16(a0, bfr, acc[0][nt]);
            acc[1][nt] = wmma_bf16(a1, bfr, acc[1][nt]);
        }
    }

#pragma unroll
    for (int mi = 0; mi < 2; ++mi) {
#pragma unroll
        for (int nt = 0; nt < 4; ++nt) {
            int ncol = nb + nt * 16 + lm;             // lane = column N
            float bv = bias[ncol];
            int h = ncol >> 6;
            int d = ncol & 63;
#pragma unroll
            for (int j = 0; j < 8; ++j) {             // VGPR j = row M (+8 hi lanes)
                int m  = mt * 32 + mi * 16 + koff + j;
                int bi = m >> 11;
                int s  = m & (SEQ - 1);
                float val = acc[mi][nt][j] + bv;
                size_t off = vmode
                    ? (((size_t)(bi * NHEAD + h) * DKH + d) * SEQ + s)
                    : (((size_t)(bi * NHEAD + h) * SEQ + s) * DKH + d);
                Y[off] = (__bf16)val;
            }
        }
    }
}

// ---------------------------------------------------- flash attention
// One wave handles (b, h, 32 queries).  Key chunks of 32 are TDM-DMA'd into
// double-buffered LDS tiles; compute overlaps the next chunk's DMA.
//   S^T(16key x 16q) = K_chunk * Q^T      (2 WMMA x 2 key tiles x 2 q tiles)
//   online softmax fp32 (per-lane reduce + shfl_xor(16))
//   ctx^T(64d x 16q) += V^T * P^T         (4 WMMA x 2 q tiles)
__global__ void flash_attn_kernel(const __bf16* __restrict__ Qb, const __bf16* __restrict__ Kb,
                                  const __bf16* __restrict__ Vt, const int* __restrict__ mask,
                                  __bf16* __restrict__ ctx) {
    __shared__ __bf16 ldsK[2][32 * DKH];   // 32 key rows x 64 d        (4 KiB each)
    __shared__ __bf16 ldsV[2][DKH * 32];   // 64 d rows  x 32 keys      (4 KiB each)

    int lane = threadIdx.x;
    int koff = (lane >> 4) * 8;
    int lm   = lane & 15;
    int qg = blockIdx.x;   // 32-query group
    int h  = blockIdx.y;
    int b  = blockIdx.z;

    const __bf16* Qh = Qb + (size_t)(b * NHEAD + h) * SEQ * DKH;
    const __bf16* Kh = Kb + (size_t)(b * NHEAD + h) * SEQ * DKH;
    const __bf16* Vh = Vt + (size_t)(b * NHEAD + h) * DKH * SEQ;
    const int* mk = mask + (size_t)b * SEQ;

    // Q^T B-fragments (lane = query column), loaded once.
    v16bf qf[2][2];
#pragma unroll
    for (int u = 0; u < 2; ++u) {
        const __bf16* Qr = Qh + (size_t)(qg * 32 + u * 16 + lm) * DKH;
        qf[u][0] = load_frag_row(Qr, 0, koff);
        qf[u][1] = load_frag_row(Qr, 32, koff);
    }

    v8f z = {};
    v8f acc[2][4];
#pragma unroll
    for (int u = 0; u < 2; ++u)
#pragma unroll
        for (int dt = 0; dt < 4; ++dt) acc[u][dt] = z;
    float m_run[2] = {-INFINITY, -INFINITY};
    float l_run[2] = {0.0f, 0.0f};
    const float scale = 0.125f;   // 1/sqrt(64)

    unsigned int lK[2] = { lds_addr_of(&ldsK[0][0]), lds_addr_of(&ldsK[1][0]) };
    unsigned int lV[2] = { lds_addr_of(&ldsV[0][0]), lds_addr_of(&ldsV[1][0]) };

    // prologue: DMA chunk 0 into buffer 0
    // K tensor [S, dk]: tile 64(d) x 32(keys);  V^T tensor [dk, S]: tile 32(keys) x 64(d)
    tdm_load_2d(lK[0], Kh,            DKH, SEQ, DKH, 32, DKH);
    tdm_load_2d(lV[0], Vh,            SEQ, DKH, 32, DKH, SEQ);

    for (int kc = 0; kc < SEQ; kc += 32) {
        int buf = (kc >> 5) & 1;
        if (kc + 32 < SEQ) {   // uniform branch: prefetch next chunk, wait for current
            tdm_load_2d(lK[buf ^ 1], Kh + (size_t)(kc + 32) * DKH, DKH, SEQ, DKH, 32, DKH);
            tdm_load_2d(lV[buf ^ 1], Vh + (kc + 32),               SEQ, DKH, 32, DKH, SEQ);
            __builtin_amdgcn_s_wait_tensorcnt(2);
        } else {
            __builtin_amdgcn_s_wait_tensorcnt(0);
        }
        asm volatile("" ::: "memory");

        const __bf16* Kt = &ldsK[buf][0];
        const __bf16* Vtile = &ldsV[buf][0];

        // K A-fragments (lane = key row), shared by both query tiles
        v16bf kf[2][2];
#pragma unroll
        for (int t = 0; t < 2; ++t) {
            const __bf16* Kr = Kt + (t * 16 + lm) * DKH;
            kf[t][0] = load_frag_row(Kr, 0, koff);
            kf[t][1] = load_frag_row(Kr, 32, koff);
        }
        // V^T A-fragments (lane = d row), shared by both query tiles
        v16bf vf[4];
#pragma unroll
        for (int dt = 0; dt < 4; ++dt)
            vf[dt] = load_frag_row(Vtile + (dt * 16 + lm) * 32, 0, koff);

#pragma unroll
        for (int u = 0; u < 2; ++u) {
            v8f st[2];
#pragma unroll
            for (int t = 0; t < 2; ++t) {
                v8f c = z;
                c = wmma_bf16(kf[t][0], qf[u][0], c);
                c = wmma_bf16(kf[t][1], qf[u][1], c);
                st[t] = c;
            }

            float pm[16];
            float mx = m_run[u];
#pragma unroll
            for (int t = 0; t < 2; ++t) {
#pragma unroll
                for (int j = 0; j < 8; ++j) {
                    int key = kc + t * 16 + koff + j;   // D elem j -> key row j(+8)
                    float s = st[t][j] * scale;
                    if (mk[key] == 0) s = -1.0e9f;
                    pm[t * 8 + j] = s;
                    mx = fmaxf(mx, s);
                }
            }
            mx = fmaxf(mx, __shfl_xor(mx, 16, 32));
            float alpha = (m_run[u] == -INFINITY) ? 0.0f : __expf(m_run[u] - mx);
            m_run[u] = mx;

            float psum = 0.0f;
            v16bf pf;                       // S^T D layout == P^T B layout
#pragma unroll
            for (int e = 0; e < 16; ++e) {
                float p = __expf(pm[e] - mx);
                psum += p;
                pf[e] = (__bf16)p;
            }
            psum += __shfl_xor(psum, 16, 32);
            l_run[u] = l_run[u] * alpha + psum;

#pragma unroll
            for (int dt = 0; dt < 4; ++dt)
#pragma unroll
                for (int j = 0; j < 8; ++j) acc[u][dt][j] *= alpha;

#pragma unroll
            for (int dt = 0; dt < 4; ++dt)
                acc[u][dt] = wmma_bf16(vf[dt], pf, acc[u][dt]);
        }
    }

    // normalize and store ctx rows (bf16, row-major [B*S, DIM]) for out-proj
#pragma unroll
    for (int u = 0; u < 2; ++u) {
        float inv = 1.0f / l_run[u];
        int trow = b * SEQ + qg * 32 + u * 16 + lm;     // ctx^T D: lane = query
        bf16x2* Cr = (bf16x2*)(ctx + (size_t)trow * DIM_ + h * DKH);
#pragma unroll
        for (int dt = 0; dt < 4; ++dt) {
#pragma unroll
            for (int p = 0; p < 4; ++p) {
                bf16x2 v = { (__bf16)(acc[u][dt][2 * p] * inv),
                             (__bf16)(acc[u][dt][2 * p + 1] * inv) };
                Cr[(dt * 16 + koff) / 2 + p] = v;       // d = dt*16 + koff + {2p,2p+1}
            }
        }
    }
}

// ------------------------------------------------ output projection (f32 out)
__global__ void out_gemm_kernel(const __bf16* __restrict__ X, const __bf16* __restrict__ W,
                                const float* __restrict__ bias, float* __restrict__ out) {
    int lane = threadIdx.x;
    int koff = (lane >> 4) * 8;
    int lm   = lane & 15;
    int mt   = blockIdx.x;
    int nb   = blockIdx.y * 64;

    const __bf16* Xr0 = X + (size_t)(mt * 32 + lm) * DIM_;
    const __bf16* Xr1 = X + (size_t)(mt * 32 + 16 + lm) * DIM_;
    v8f z = {};
    v8f acc[2][4];
#pragma unroll
    for (int mi = 0; mi < 2; ++mi)
#pragma unroll
        for (int nt = 0; nt < 4; ++nt) acc[mi][nt] = z;

    for (int kb = 0; kb < DIM_; kb += 32) {
        v16bf a0 = load_frag_row(Xr0, kb, koff);
        v16bf a1 = load_frag_row(Xr1, kb, koff);
#pragma unroll
        for (int nt = 0; nt < 4; ++nt) {
            v16bf bfr = load_frag_row(W + (size_t)(nb + nt * 16 + lm) * DIM_, kb, koff);
            acc[0][nt] = wmma_bf16(a0, bfr, acc[0][nt]);
            acc[1][nt] = wmma_bf16(a1, bfr, acc[1][nt]);
        }
    }

#pragma unroll
    for (int mi = 0; mi < 2; ++mi) {
#pragma unroll
        for (int nt = 0; nt < 4; ++nt) {
            int ncol = nb + nt * 16 + lm;
            float bv = bias[ncol];
#pragma unroll
            for (int j = 0; j < 8; ++j) {
                int m = mt * 32 + mi * 16 + koff + j;
                out[(size_t)m * DIM_ + ncol] = acc[mi][nt][j] + bv;
            }
        }
    }
}

// -----------------------------------------------------------------------------
extern "C" void kernel_launch(void* const* d_in, const int* in_sizes, int n_in,
                              void* d_out, int out_size, void* d_ws, size_t ws_size,
                              hipStream_t stream) {
    (void)in_sizes; (void)n_in; (void)out_size; (void)ws_size;

    const float* query = (const float*)d_in[0];
    const float* key_  = (const float*)d_in[1];
    const float* value = (const float*)d_in[2];
    const int*   mask  = (const int*)d_in[3];
    const float* Wq = (const float*)d_in[4];
    const float* bq = (const float*)d_in[5];
    const float* Wk = (const float*)d_in[6];
    const float* bk = (const float*)d_in[7];
    const float* Wv = (const float*)d_in[8];
    const float* bv = (const float*)d_in[9];
    const float* Wo = (const float*)d_in[10];
    const float* bo = (const float*)d_in[11];
    float* out = (float*)d_out;

    // workspace layout (64 MiB total)
    char* ws = (char*)d_ws;
    const size_t SZ_X = (size_t)MTOT * DIM_ * sizeof(__bf16);  // 8 MiB
    const size_t SZ_W = (size_t)DIM_ * DIM_ * sizeof(__bf16);  // 2 MiB
    __bf16* Xq  = (__bf16*)(ws);
    __bf16* Xk  = (__bf16*)(ws + SZ_X);
    __bf16* Xv  = (__bf16*)(ws + 2 * SZ_X);
    __bf16* Wqb = (__bf16*)(ws + 3 * SZ_X);
    __bf16* Wkb = (__bf16*)(ws + 3 * SZ_X + SZ_W);
    __bf16* Wvb = (__bf16*)(ws + 3 * SZ_X + 2 * SZ_W);
    __bf16* Wob = (__bf16*)(ws + 3 * SZ_X + 3 * SZ_W);
    __bf16* Qbf = (__bf16*)(ws + 3 * SZ_X + 4 * SZ_W);
    __bf16* Kbf = (__bf16*)(ws + 4 * SZ_X + 4 * SZ_W);
    __bf16* Vtb = (__bf16*)(ws + 5 * SZ_X + 4 * SZ_W);
    __bf16* Ctx = (__bf16*)(ws + 6 * SZ_X + 4 * SZ_W);

    const int nx = MTOT * DIM_;
    const int nw = DIM_ * DIM_;
    cvt_f32_bf16<<<(nx + 255) / 256, 256, 0, stream>>>(query, Xq, nx);
    cvt_f32_bf16<<<(nx + 255) / 256, 256, 0, stream>>>(key_,  Xk, nx);
    cvt_f32_bf16<<<(nx + 255) / 256, 256, 0, stream>>>(value, Xv, nx);
    cvt_f32_bf16<<<(nw + 255) / 256, 256, 0, stream>>>(Wq, Wqb, nw);
    cvt_f32_bf16<<<(nw + 255) / 256, 256, 0, stream>>>(Wk, Wkb, nw);
    cvt_f32_bf16<<<(nw + 255) / 256, 256, 0, stream>>>(Wv, Wvb, nw);
    cvt_f32_bf16<<<(nw + 255) / 256, 256, 0, stream>>>(Wo, Wob, nw);

    dim3 pg(MTOT / 32, DIM_ / 64);   // 128 x 16 waves
    proj_gemm_kernel<<<pg, 32, 0, stream>>>(Xq, Wqb, bq, Qbf, 0);
    proj_gemm_kernel<<<pg, 32, 0, stream>>>(Xk, Wkb, bk, Kbf, 0);
    proj_gemm_kernel<<<pg, 32, 0, stream>>>(Xv, Wvb, bv, Vtb, 1);

    flash_attn_kernel<<<dim3(SEQ / 32, NHEAD, BATCH), 32, 0, stream>>>(Qbf, Kbf, Vtb, mask, Ctx);

    out_gemm_kernel<<<pg, 32, 0, stream>>>(Ctx, Wob, bo, out);
}
